// Attention_14783277433146
// MI455X (gfx1250) — matrix operand order repaired and checked by reading, not compiled, not run on hardware
//
#include <hip/hip_runtime.h>

typedef __attribute__((ext_vector_type(16))) _Float16 v16h;
typedef __attribute__((ext_vector_type(8)))  _Float16 v8h;
typedef __attribute__((ext_vector_type(4)))  _Float16 v4h;
typedef __attribute__((ext_vector_type(8)))  float    v8f;
typedef __attribute__((ext_vector_type(4)))  int      v4i;

#define WMMA_F16(a, b, c) \
    __builtin_amdgcn_wmma_f32_16x16x32_f16(false, (a), false, (b), (short)0, (c), false, false)

#define BATCH 8
#define SEQ   1024
#define DIM   1024
#define HEADS 16
#define HDIM  64
#define SCALE 0.125f   // 64^-0.5

__device__ __forceinline__ v16h cat8(v8h lo, v8h hi) {
    return __builtin_shufflevector(lo, hi, 0,1,2,3,4,5,6,7,8,9,10,11,12,13,14,15);
}

// ---------------------------------------------------------------------------
// Async global->LDS staging (CDNA5 GLOBAL_LOAD_ASYNC_TO_LDS_B128, ASYNCcnt)
// with a VGPR-bounce fallback if the builtin is not declared.
// Probed signature: (global v4i*, local v4i*, imm offset, imm cpol).
// ---------------------------------------------------------------------------
#if __has_builtin(__builtin_amdgcn_global_load_async_to_lds_b128)
#define HAVE_ASYNC_LDS 1
typedef __attribute__((address_space(1))) v4i g_v4i;
typedef __attribute__((address_space(3))) v4i l_v4i;
__device__ __forceinline__ void stage16(_Float16* dst, const _Float16* src) {
    __builtin_amdgcn_global_load_async_to_lds_b128(
        (g_v4i*)src, (l_v4i*)dst, /*offset=*/0, /*cpol=*/0);
}
__device__ __forceinline__ void async_wait() {
#if __has_builtin(__builtin_amdgcn_s_wait_asynccnt)
    __builtin_amdgcn_s_wait_asynccnt(0);
#else
    asm volatile("s_wait_asynccnt 0x0" ::: "memory");
#endif
}
#else
#define HAVE_ASYNC_LDS 0
__device__ __forceinline__ void stage16(_Float16* dst, const _Float16* src) {
    *(v8h*)dst = *(const v8h*)src;
}
__device__ __forceinline__ void async_wait() {}
#endif

// ---------------------------------------------------------------------------
// Stage 0a: f32 -> f16 elementwise convert (n multiple of 4)
// ---------------------------------------------------------------------------
__global__ __launch_bounds__(256) void cvt_f32_f16(const float* __restrict__ in,
                                                   _Float16* __restrict__ out, int n) {
    int i = (blockIdx.x * 256 + threadIdx.x) * 4;
    if (i + 3 < n) {
        float4 f = *(const float4*)(in + i);
        v4h h;
        h[0] = (_Float16)f.x; h[1] = (_Float16)f.y;
        h[2] = (_Float16)f.z; h[3] = (_Float16)f.w;
        *(v4h*)(out + i) = h;
    }
}

// ---------------------------------------------------------------------------
// Stage 0b: transpose + convert: out[c][r] = (f16) in[r][c]
// ---------------------------------------------------------------------------
__global__ __launch_bounds__(256) void cvt_transpose_f16(const float* __restrict__ in,
                                                         _Float16* __restrict__ out,
                                                         int rows, int cols) {
    long idx = (long)blockIdx.x * 256 + threadIdx.x;
    if (idx < (long)rows * cols) {
        int r = (int)(idx % rows);
        int c = (int)(idx / rows);
        out[(size_t)c * rows + r] = (_Float16)in[(size_t)r * cols + c];
    }
}

// ---------------------------------------------------------------------------
// Stage 1: qkv = x @ w_qkv. Block = 8 waves = 256(M) x 64(N); each wave does
// 32x64 (8 accumulators -> 8 WMMAs per staged B tile). B tile (64x32 halves)
// is double-buffered in LDS, staged with async global->LDS loads, ONE barrier
// per K-step. N-tile is 64-aligned so (which,h) are blockIdx-scalar.
// ---------------------------------------------------------------------------
__global__ __launch_bounds__(256) void gemm_qkv(const _Float16* __restrict__ A,
                                                const _Float16* __restrict__ Bt,
                                                _Float16* __restrict__ Q,
                                                _Float16* __restrict__ K,
                                                _Float16* __restrict__ Vt) {
    __shared__ _Float16 ldsB[2][64][32];            // 8 KB double-buffered
    const int NTB = (3 * DIM) / 64;                 // 48 col-blocks
    const int tmb = blockIdx.x / NTB;               // M block (256 rows)
    const int tnb = blockIdx.x % NTB;               // N block (64 cols)
    const int wslot = threadIdx.x >> 5;
    const int lane  = threadIdx.x & 31;
    const int half  = lane >> 4;
    const int lid   = lane & 15;

    const int m0 = tmb * 256 + wslot * 32;
    const _Float16* arow0 = A + (size_t)(m0 + lid) * DIM;
    const _Float16* arow1 = A + (size_t)(m0 + 16 + lid) * DIM;
    const _Float16* bbas  = Bt + (size_t)(tnb * 64) * DIM;
    const int ln = threadIdx.x >> 2;                // 0..63   (B row)
    const int lk = (threadIdx.x & 3) * 8;           // 0,8,16,24

    v8f acc[8] = {};
    int buf = 0;
    stage16(&ldsB[0][ln][lk], bbas + (size_t)ln * DIM + lk);
    for (int k0 = 0; k0 < DIM; k0 += 32) {
        async_wait();
        __syncthreads();
        if (k0 + 32 < DIM)
            stage16(&ldsB[buf ^ 1][ln][lk], bbas + (size_t)ln * DIM + k0 + 32 + lk);
        __builtin_prefetch(arow0 + k0 + 64, 0, 1);

        v16h a0 = cat8(*(const v8h*)(arow0 + k0 + half * 16),
                       *(const v8h*)(arow0 + k0 + 8 + half * 16));
        v16h a1 = cat8(*(const v8h*)(arow1 + k0 + half * 16),
                       *(const v8h*)(arow1 + k0 + 8 + half * 16));
#pragma unroll
        for (int t = 0; t < 4; t++) {
            v16h b = *(const v16h*)&ldsB[buf][t * 16 + lid][half * 16];
            acc[t]     = WMMA_F16(a0, b, acc[t]);
            acc[4 + t] = WMMA_F16(a1, b, acc[4 + t]);
        }
        buf ^= 1;
    }

    const int c0    = tnb * 64;                     // scalar
    const int which = c0 >> 10;                     // 0=q 1=k 2=v (scalar)
    const int hh    = (c0 & 1023) >> 6;             // head (scalar)
#pragma unroll
    for (int ms = 0; ms < 2; ms++) {
#pragma unroll
        for (int t = 0; t < 4; t++) {
            const int dd = t * 16 + lid;            // 0..63 within head
#pragma unroll
            for (int v = 0; v < 8; v++) {
                int r  = m0 + ms * 16 + v + half * 8;
                int b  = r >> 10;
                int n  = r & 1023;
                int bh = b * HEADS + hh;
                float val = acc[ms * 4 + t][v];
                if (which == 0)
                    Q[((size_t)bh * SEQ + n) * HDIM + dd] = (_Float16)(val * SCALE);
                else if (which == 1)
                    K[((size_t)bh * SEQ + n) * HDIM + dd] = (_Float16)val;
                else
                    Vt[((size_t)bh * HDIM + dd) * SEQ + n] = (_Float16)val;
            }
        }
    }
}

// ---------------------------------------------------------------------------
// Stage 2: flash attention. Block = 8 waves on the SAME (b,h). K/V 32-key
// tiles double-buffered in LDS (async staged), one barrier per step. Per wave:
// S = Q K^T (4 WMMAs), online softmax (16-lane shfl reductions), P re-laid out
// C-frag -> A-frag via wave-private LDS, O += P V (4 WMMAs).
// ---------------------------------------------------------------------------
__global__ __launch_bounds__(256) void attn_kernel(const _Float16* __restrict__ Q,
                                                   const _Float16* __restrict__ K,
                                                   const _Float16* __restrict__ Vt,
                                                   _Float16* __restrict__ Ao) {
    __shared__ _Float16 ldsK[2][32][64];            // 8 KB
    __shared__ _Float16 ldsV[2][64][32];            // 8 KB
    __shared__ _Float16 ldsP[8][16][32];            // 8 KB per-wave P staging
    const int wslot = threadIdx.x >> 5;
    const int lane  = threadIdx.x & 31;
    const int half  = lane >> 4;
    const int lid   = lane & 15;

    const int bh = blockIdx.x >> 3;                 // scalar
    const int qb = (blockIdx.x & 7) * 8 + wslot;    // query block (16 rows)

    const _Float16* qbase = Q  + ((size_t)bh * SEQ + qb * 16) * HDIM;
    const _Float16* kbase = K  + (size_t)bh * SEQ * HDIM;
    const _Float16* vbase = Vt + (size_t)bh * HDIM * SEQ;

    // Q A-fragments resident in registers
    const _Float16* qr = qbase + lid * HDIM;
    v16h qa0 = cat8(*(const v8h*)(qr +      half * 16), *(const v8h*)(qr + 8 + half * 16));
    v16h qa1 = cat8(*(const v8h*)(qr + 32 + half * 16), *(const v8h*)(qr + 40 + half * 16));

    const int kj = threadIdx.x >> 3;                // 0..31 (key row)
    const int kd = (threadIdx.x & 7) * 8;           // 0..56 (d offset)
    const int vd = threadIdx.x >> 2;                // 0..63 (d row)
    const int vj = (threadIdx.x & 3) * 8;           // 0..24 (key offset)

    float mrow[8], lrow[8];
#pragma unroll
    for (int v = 0; v < 8; v++) { mrow[v] = -3.0e38f; lrow[v] = 0.0f; }
    v8f o0 = {}, o1 = {}, o2 = {}, o3 = {};

    int buf = 0;
    stage16(&ldsK[0][kj][kd], kbase + (size_t)kj * HDIM + kd);
    stage16(&ldsV[0][vd][vj], vbase + (size_t)vd * SEQ + vj);
    for (int j0 = 0; j0 < SEQ; j0 += 32) {
        async_wait();
        __syncthreads();
        if (j0 + 32 < SEQ) {
            stage16(&ldsK[buf ^ 1][kj][kd], kbase + (size_t)(j0 + 32 + kj) * HDIM + kd);
            stage16(&ldsV[buf ^ 1][vd][vj], vbase + (size_t)vd * SEQ + j0 + 32 + vj);
        }

        // ---- S tiles: keys [j0,j0+16) and [j0+16,j0+32)
        v16h kb0 = *(const v16h*)&ldsK[buf][lid][half * 16];
        v16h kb1 = *(const v16h*)&ldsK[buf][lid][32 + half * 16];
        v16h kb2 = *(const v16h*)&ldsK[buf][16 + lid][half * 16];
        v16h kb3 = *(const v16h*)&ldsK[buf][16 + lid][32 + half * 16];
        v8f s0 = {}, s1 = {};
        s0 = WMMA_F16(qa0, kb0, s0);
        s0 = WMMA_F16(qa1, kb1, s0);
        s1 = WMMA_F16(qa0, kb2, s1);
        s1 = WMMA_F16(qa1, kb3, s1);

        // ---- online softmax across the 16-lane column groups
#pragma unroll
        for (int v = 0; v < 8; v++) {
            float bm = fmaxf(s0[v], s1[v]);
            for (int m = 8; m; m >>= 1) bm = fmaxf(bm, __shfl_xor(bm, m, 32));
            float nm = fmaxf(mrow[v], bm);
            float f  = __expf(mrow[v] - nm);
            float p0 = __expf(s0[v] - nm);
            float p1 = __expf(s1[v] - nm);
            float rs = p0 + p1;
            for (int m = 8; m; m >>= 1) rs += __shfl_xor(rs, m, 32);
            lrow[v] = lrow[v] * f + rs;
            mrow[v] = nm;
            o0[v] *= f; o1[v] *= f; o2[v] *= f; o3[v] *= f;
            int rr = v + half * 8;
            ldsP[wslot][rr][lid]      = (_Float16)p0;
            ldsP[wslot][rr][16 + lid] = (_Float16)p1;
        }

        // ---- C-layout -> A-layout via wave-private LDS (in-order DS)
        const _Float16* pr = &ldsP[wslot][lid][0];
        v16h pa = cat8(*(const v8h*)(pr + half * 16), *(const v8h*)(pr + 8 + half * 16));

        // ---- O += P(16x32) @ V(32x64)
        v16h vb0 = *(const v16h*)&ldsV[buf][lid][half * 16];
        v16h vb1 = *(const v16h*)&ldsV[buf][16 + lid][half * 16];
        v16h vb2 = *(const v16h*)&ldsV[buf][32 + lid][half * 16];
        v16h vb3 = *(const v16h*)&ldsV[buf][48 + lid][half * 16];
        o0 = WMMA_F16(pa, vb0, o0);
        o1 = WMMA_F16(pa, vb1, o1);
        o2 = WMMA_F16(pa, vb2, o2);
        o3 = WMMA_F16(pa, vb3, o3);
        buf ^= 1;
    }

    // ---- normalize + store to Ao[b, n, h*64 + d]
    const int b  = bh >> 4;
    const int hh = bh & 15;
#pragma unroll
    for (int v = 0; v < 8; v++) {
        float inv = 1.0f / lrow[v];
        size_t row  = (size_t)b * SEQ + qb * 16 + v + half * 8;
        size_t base = row * DIM + hh * HDIM + lid;
        Ao[base +  0] = (_Float16)(o0[v] * inv);
        Ao[base + 16] = (_Float16)(o1[v] * inv);
        Ao[base + 32] = (_Float16)(o2[v] * inv);
        Ao[base + 48] = (_Float16)(o3[v] * inv);
    }
}

// ---------------------------------------------------------------------------
// Stage 3: out = Ao @ w_proj + b_proj (f32 out). Same structure as gemm_qkv.
// ---------------------------------------------------------------------------
__global__ __launch_bounds__(256) void gemm_proj(const _Float16* __restrict__ A,
                                                 const _Float16* __restrict__ Bt,
                                                 const float* __restrict__ bias,
                                                 float* __restrict__ out) {
    __shared__ _Float16 ldsB[2][64][32];
    const int NTB = DIM / 64;                       // 16 col-blocks
    const int tmb = blockIdx.x / NTB;
    const int tnb = blockIdx.x % NTB;
    const int wslot = threadIdx.x >> 5;
    const int lane  = threadIdx.x & 31;
    const int half  = lane >> 4;
    const int lid   = lane & 15;

    const int m0 = tmb * 256 + wslot * 32;
    const _Float16* arow0 = A + (size_t)(m0 + lid) * DIM;
    const _Float16* arow1 = A + (size_t)(m0 + 16 + lid) * DIM;
    const _Float16* bbas  = Bt + (size_t)(tnb * 64) * DIM;
    const int ln = threadIdx.x >> 2;
    const int lk = (threadIdx.x & 3) * 8;

    v8f acc[8] = {};
    int buf = 0;
    stage16(&ldsB[0][ln][lk], bbas + (size_t)ln * DIM + lk);
    for (int k0 = 0; k0 < DIM; k0 += 32) {
        async_wait();
        __syncthreads();
        if (k0 + 32 < DIM)
            stage16(&ldsB[buf ^ 1][ln][lk], bbas + (size_t)ln * DIM + k0 + 32 + lk);
        __builtin_prefetch(arow0 + k0 + 64, 0, 1);

        v16h a0 = cat8(*(const v8h*)(arow0 + k0 + half * 16),
                       *(const v8h*)(arow0 + k0 + 8 + half * 16));
        v16h a1 = cat8(*(const v8h*)(arow1 + k0 + half * 16),
                       *(const v8h*)(arow1 + k0 + 8 + half * 16));
#pragma unroll
        for (int t = 0; t < 4; t++) {
            v16h b = *(const v16h*)&ldsB[buf][t * 16 + lid][half * 16];
            acc[t]     = WMMA_F16(a0, b, acc[t]);
            acc[4 + t] = WMMA_F16(a1, b, acc[4 + t]);
        }
        buf ^= 1;
    }

#pragma unroll
    for (int ms = 0; ms < 2; ms++) {
#pragma unroll
        for (int t = 0; t < 4; t++) {
            int c = tnb * 64 + t * 16 + lid;
            float bv = bias[c];
#pragma unroll
            for (int v = 0; v < 8; v++) {
                size_t r = (size_t)(m0 + ms * 16 + v + half * 8);
                out[r * DIM + c] = acc[ms * 4 + t][v] + bv;
            }
        }
    }
}

// ---------------------------------------------------------------------------
extern "C" void kernel_launch(void* const* d_in, const int* in_sizes, int n_in,
                              void* d_out, int out_size, void* d_ws, size_t ws_size,
                              hipStream_t stream) {
    const float* x      = (const float*)d_in[0];   // [8,1024,1024]
    const float* w_qkv  = (const float*)d_in[1];   // [1024,3072]
    const float* w_proj = (const float*)d_in[2];   // [1024,1024]
    const float* b_proj = (const float*)d_in[3];   // [1024]
    float* out = (float*)d_out;                    // [8,1024,1024]

    const size_t NROWS = (size_t)BATCH * SEQ;      // 8192
    _Float16* ws     = (_Float16*)d_ws;
    _Float16* xh     = ws;                                   // 8M halves
    _Float16* wqkvT  = xh     + NROWS * DIM;                 // 3M
    _Float16* wprojT = wqkvT  + (size_t)3 * DIM * DIM;       // 1M
    _Float16* Qb     = wprojT + (size_t)DIM * DIM;           // 8M
    _Float16* Kb     = Qb     + NROWS * DIM;                 // 8M
    _Float16* Vt     = Kb     + NROWS * DIM;                 // 8M
    _Float16* Ao     = Vt     + NROWS * DIM;                 // 8M

    // Stage 0: precision / layout transforms
    cvt_f32_f16<<<(NROWS * DIM) / 1024, 256, 0, stream>>>(x, xh, (int)(NROWS * DIM));
    cvt_transpose_f16<<<(3 * DIM * DIM) / 256, 256, 0, stream>>>(w_qkv, wqkvT, DIM, 3 * DIM);
    cvt_transpose_f16<<<(DIM * DIM) / 256, 256, 0, stream>>>(w_proj, wprojT, DIM, DIM);

    // Stage 1: QKV GEMM (8192 x 3072 x 1024) -> Q, K, Vt
    gemm_qkv<<<(8192 / 256) * (3 * DIM / 64), 256, 0, stream>>>(xh, wqkvT, Qb, Kb, Vt);

    // Stage 2: attention (128 bh, 8 blocks each of 8 waves x 16 rows)
    attn_kernel<<<BATCH * HEADS * 8, 256, 0, stream>>>(Qb, Kb, Vt, Ao);

    // Stage 3: proj GEMM (8192 x 1024 x 1024) + bias -> f32 out
    gemm_proj<<<(8192 / 256) * (DIM / 64), 256, 0, stream>>>(Ao, wprojT, b_proj, out);
}
